// NTMCell_88862873354490
// MI455X (gfx1250) — matrix operand order repaired and checked by reading, not compiled, not run on hardware
//
#include <hip/hip_runtime.h>

// ---------------- problem constants ----------------
#define B_    256
#define I_    256
#define C_    1024
#define N_    512
#define M_    256
#define KPAD  1568      // 513 + 1024 = 1537 padded to multiple of 32
#define NZ    4096      // 4*C
#define NST   2048      // 4*N
#define NP    832       // 262 + 262 + 256 + 4 + 1 = 785 padded to multiple of 64

// d_out layout: out_state[:, -1] (256) | reads (65536) | mem_new (33554432) | gate (1024)
#define OUT_LAST_OFF 0
#define READS_OFF    256
#define MEM_OFF      65792
#define GATE_OFF     33620224

typedef __bf16 v16bf __attribute__((ext_vector_type(16)));
typedef __bf16 v8bf  __attribute__((ext_vector_type(8)));
typedef float  v8f   __attribute__((ext_vector_type(8)));

#if __has_builtin(__builtin_amdgcn_sched_barrier)
#define SCHED_FENCE() __builtin_amdgcn_sched_barrier(0)
#else
#define SCHED_FENCE()
#endif

__device__ __forceinline__ float sigf(float x)      { return 1.0f / (1.0f + __expf(-x)); }
__device__ __forceinline__ float softplusf(float x) { return (x > 20.0f) ? x : log1pf(expf(x)); }

// ================= WMMA GEMM =================
// C[M,N] = bf16(A[M,K]) @ bf16frag(B[K,N]) + bias, optional tanh.
// Block = 4 waves; block tile 64x64. Wave w owns M-tile (m0 + 16w), all 4 N-tiles.
// The 32x64 B slab (4 KB) is shared by all 4 waves via LDS, double-buffered.
// Scheduling fences pin the intended pipeline:
//   [A loads + next-slab prefetch] | [8 ds_loads] | [4 WMMAs] | [commit slab] barrier
// so the global prefetch latency is covered by the WMMA batch instead of being
// exposed at the barrier (round-3 disasm showed the scheduler sinking it there).
// B is pre-swizzled into fragment order: tile (tk,tn) of 32x16 stored as 512
// contiguous bf16 in lane-major order -> per lane: two 16B vector loads.
__global__ __launch_bounds__(128)
void gemm_bf16_wmma(const __bf16* __restrict__ A, const __bf16* __restrict__ Bf,
                    const float* __restrict__ bias, float* __restrict__ C,
                    int Ncols, int K, int act)
{
    __shared__ __align__(16) __bf16 lds[2][2048];   // 2 stages x (4 tiles x 512)

    const int n0   = blockIdx.x * 64;
    const int m0   = blockIdx.y * 64;
    const int tid  = threadIdx.x;
    const int wave = tid >> 5;
    const int lane = tid & 31;
    const int half = lane >> 4;
    const int lr   = lane & 15;
    const int nt   = Ncols >> 4;       // fragment tiles along N
    const int nk   = K >> 5;           // k-steps

    v8f acc[4] = {v8f{}, v8f{}, v8f{}, v8f{}};

    const __bf16* Arow = A + (size_t)(m0 + wave * 16 + lr) * K;

    // preload stage 0: 128 threads x 16 bf16 = 2048 bf16 (one 32x64 B slab)
    {
        const __bf16* Bt = Bf + ((size_t)0 * nt + (size_t)(n0 >> 4)) * 512;
        v8bf t0 = *(const v8bf*)(Bt + tid * 16);
        v8bf t1 = *(const v8bf*)(Bt + tid * 16 + 8);
        *(v8bf*)(&lds[0][tid * 16])     = t0;
        *(v8bf*)(&lds[0][tid * 16 + 8]) = t1;
    }
    __syncthreads();

    for (int tk = 0; tk < nk - 1; ++tk) {
        const int cur = tk & 1, nxt = cur ^ 1;

        // (1) A fragment for this step (needed by the WMMAs), then the next-slab
        //     prefetch. Both issued up front; fence keeps them here.
        v8bf alo = *(const v8bf*)(Arow + tk * 32 + half * 16);
        v8bf ahi = *(const v8bf*)(Arow + tk * 32 + half * 16 + 8);
        const __bf16* Bt = Bf + ((size_t)(tk + 1) * nt + (size_t)(n0 >> 4)) * 512;
        v8bf p0 = *(const v8bf*)(Bt + tid * 16);
        v8bf p1 = *(const v8bf*)(Bt + tid * 16 + 8);
        SCHED_FENCE();

        // (2) stage all four B fragments from LDS (batched ds_loads)
        v8bf blo[4], bhi[4];
#pragma unroll
        for (int t = 0; t < 4; ++t) {
            const __bf16* bp = &lds[cur][t * 512 + lane * 16];
            blo[t] = *(const v8bf*)(bp);
            bhi[t] = *(const v8bf*)(bp + 8);
        }
        SCHED_FENCE();

        v16bf a;
#pragma unroll
        for (int i = 0; i < 8; ++i) { a[i] = alo[i]; a[8 + i] = ahi[i]; }

        // (3) back-to-back WMMAs (cover the outstanding prefetch)
#pragma unroll
        for (int t = 0; t < 4; ++t) {
            v16bf b;
#pragma unroll
            for (int i = 0; i < 8; ++i) { b[i] = blo[t][i]; b[8 + i] = bhi[t][i]; }
            acc[t] = __builtin_amdgcn_wmma_f32_16x16x32_bf16(
                false, a, false, b, (short)0, acc[t], false, false);
        }
        SCHED_FENCE();

        // (4) commit prefetched slab, flip stages
        *(v8bf*)(&lds[nxt][tid * 16])     = p0;
        *(v8bf*)(&lds[nxt][tid * 16 + 8]) = p1;
        __syncthreads();
    }

    // peeled final k-step (no prefetch)
    {
        const int tk = nk - 1;
        const int cur = tk & 1;
        v8bf alo = *(const v8bf*)(Arow + tk * 32 + half * 16);
        v8bf ahi = *(const v8bf*)(Arow + tk * 32 + half * 16 + 8);
        v8bf blo[4], bhi[4];
#pragma unroll
        for (int t = 0; t < 4; ++t) {
            const __bf16* bp = &lds[cur][t * 512 + lane * 16];
            blo[t] = *(const v8bf*)(bp);
            bhi[t] = *(const v8bf*)(bp + 8);
        }
        v16bf a;
#pragma unroll
        for (int i = 0; i < 8; ++i) { a[i] = alo[i]; a[8 + i] = ahi[i]; }
#pragma unroll
        for (int t = 0; t < 4; ++t) {
            v16bf b;
#pragma unroll
            for (int i = 0; i < 8; ++i) { b[i] = blo[t][i]; b[8 + i] = bhi[t][i]; }
            acc[t] = __builtin_amdgcn_wmma_f32_16x16x32_bf16(
                false, a, false, b, (short)0, acc[t], false, false);
        }
    }

    // epilogue: D VGPR i -> row m0 + wave*16 + i + 8*half, col n0 + t*16 + lr
#pragma unroll
    for (int t = 0; t < 4; ++t) {
        const int col = n0 + t * 16 + lr;
        const float bv = bias ? bias[col] : 0.0f;
#pragma unroll
        for (int i = 0; i < 8; ++i) {
            const int row = m0 + wave * 16 + i + 8 * half;
            float v = acc[t][i] + bv;
            if (act == 1) v = tanhf(v);
            C[(size_t)row * Ncols + col] = v;
        }
    }
}

// ================= pack kernels =================
__global__ __launch_bounds__(256)
void pack_A_lstm(const float* __restrict__ x, const float* __restrict__ err,
                 const float* __restrict__ preads, const float* __restrict__ ph,
                 __bf16* __restrict__ Abig)
{
    int idx = blockIdx.x * 256 + threadIdx.x;
    if (idx >= B_ * KPAD) return;
    int b = idx / KPAD, k = idx % KPAD;
    float v;
    if (k < 256)       v = x[(size_t)b * 256 + k];
    else if (k == 256) v = err[b];
    else if (k < 513)  v = preads[(size_t)b * 256 + (k - 257)];
    else if (k < 1537) v = ph[(size_t)b * 1024 + (k - 513)];
    else               v = 0.0f;
    Abig[idx] = (__bf16)v;
}

// fragment-order pack: idx -> (tile, lane, e); krow = tk*32 + (lane/16)*16 + e; col = tn*16 + lane%16
__global__ __launch_bounds__(256)
void packB_z(const float* __restrict__ Wx, const float* __restrict__ Wh, __bf16* __restrict__ Bf)
{
    size_t idx = (size_t)blockIdx.x * 256 + threadIdx.x;
    const size_t total = (size_t)(KPAD / 32) * (NZ / 16) * 512;
    if (idx >= total) return;
    size_t tile = idx >> 9;
    int r = (int)(idx & 511), lane = r >> 4, e = r & 15;
    int tn = (int)(tile % (NZ / 16)), tk = (int)(tile / (NZ / 16));
    int col  = tn * 16 + (lane & 15);
    int krow = tk * 32 + (lane >> 4) * 16 + e;
    float v = 0.0f;
    if (krow < 513)       v = Wx[(size_t)krow * NZ + col];
    else if (krow < 1537) v = Wh[(size_t)(krow - 513) * NZ + col];
    Bf[idx] = (__bf16)v;
}

__global__ __launch_bounds__(256)
void packB_states(const float* __restrict__ Wst, __bf16* __restrict__ Bf)
{
    size_t idx = (size_t)blockIdx.x * 256 + threadIdx.x;
    const size_t total = (size_t)(C_ / 32) * (NST / 16) * 512;
    if (idx >= total) return;
    size_t tile = idx >> 9;
    int r = (int)(idx & 511), lane = r >> 4, e = r & 15;
    int tn = (int)(tile % (NST / 16)), tk = (int)(tile / (NST / 16));
    int col  = tn * 16 + (lane & 15);          // j = k*512 + n
    int krow = tk * 32 + (lane >> 4) * 16 + e; // c
    float v = Wst[(size_t)(col >> 9) * (C_ * 512) + (size_t)krow * 512 + (col & 511)];
    Bf[idx] = (__bf16)v;
}

__global__ __launch_bounds__(256)
void packB_head(const float* __restrict__ Wr, const float* __restrict__ Ww,
                const float* __restrict__ We, const float* __restrict__ Wg,
                const float* __restrict__ Wc, __bf16* __restrict__ Bf)
{
    size_t idx = (size_t)blockIdx.x * 256 + threadIdx.x;
    const size_t total = (size_t)(C_ / 32) * (NP / 16) * 512;
    if (idx >= total) return;
    size_t tile = idx >> 9;
    int r = (int)(idx & 511), lane = r >> 4, e = r & 15;
    int tn = (int)(tile % (NP / 16)), tk = (int)(tile / (NP / 16));
    int j = tn * 16 + (lane & 15);
    int c = tk * 32 + (lane >> 4) * 16 + e;
    float v = 0.0f;
    if (j < 262)       v = Wr[(size_t)c * 262 + j];
    else if (j < 524)  v = Ww[(size_t)c * 262 + (j - 262)];
    else if (j < 780)  v = We[(size_t)c * 256 + (j - 524)];
    else if (j < 784)  v = Wg[(size_t)c * 4 + (j - 780)];
    else if (j == 784) v = Wc[c];
    Bf[idx] = (__bf16)v;
}

__global__ __launch_bounds__(256)
void packB_add(const float* __restrict__ Wa, __bf16* __restrict__ Bf)
{
    size_t idx = (size_t)blockIdx.x * 256 + threadIdx.x;
    const size_t total = (size_t)(N_ / 32) * (M_ / 16) * 512;
    if (idx >= total) return;
    size_t tile = idx >> 9;
    int r = (int)(idx & 511), lane = r >> 4, e = r & 15;
    int tn = (int)(tile % (M_ / 16)), tk = (int)(tile / (M_ / 16));
    int col  = tn * 16 + (lane & 15);
    int krow = tk * 32 + (lane >> 4) * 16 + e;
    Bf[idx] = (__bf16)Wa[(size_t)krow * M_ + col];
}

__global__ __launch_bounds__(256)
void pack_bhead(const float* __restrict__ br, const float* __restrict__ bw,
                const float* __restrict__ be, const float* __restrict__ bg,
                const float* __restrict__ bc, float* __restrict__ bhead)
{
    int j = blockIdx.x * 256 + threadIdx.x;
    if (j >= NP) return;
    float v = 0.0f;
    if (j < 262)       v = br[j];
    else if (j < 524)  v = bw[j - 262];
    else if (j < 780)  v = be[j - 524];
    else if (j < 784)  v = bg[j - 780];
    else if (j == 784) v = bc[0];
    bhead[j] = v;
}

// ================= LSTM elementwise =================
__global__ __launch_bounds__(256)
void lstm_ew(const float* __restrict__ z, const float* __restrict__ prev_c,
             __bf16* __restrict__ Ah)
{
    int idx = blockIdx.x * 256 + threadIdx.x;
    if (idx >= B_ * C_) return;
    int b = idx / C_, j = idx % C_;
    const float* zr = z + (size_t)b * NZ;
    float iv = zr[j], fv = zr[C_ + j], gv = zr[2 * C_ + j], ov = zr[3 * C_ + j];
    float cn = sigf(fv) * prev_c[idx] + sigf(iv) * tanhf(gv);
    float h  = sigf(ov) * tanhf(cn);
    Ah[idx] = (__bf16)h;
}

// ================= gate blend + out_state =================
__global__ __launch_bounds__(512)
void gate_state_kernel(const float* __restrict__ P, const float* __restrict__ prev_gate,
                       const float* __restrict__ S, __bf16* __restrict__ Aout,
                       float* __restrict__ out)
{
    int b = blockIdx.x;
    int n = threadIdx.x;
    const float* Pb = P + (size_t)b * NP;
    float g0 = Pb[780], g1 = Pb[781], g2 = Pb[782], g3 = Pb[783];
    float mx = fmaxf(fmaxf(g0, g1), fmaxf(g2, g3));
    float e0 = expf(g0 - mx), e1 = expf(g1 - mx), e2 = expf(g2 - mx), e3 = expf(g3 - mx);
    float es = e0 + e1 + e2 + e3;
    float crol = sigf(Pb[784]);
    const float* pg = prev_gate + (size_t)b * 4;
    float gm0 = (e0 / es) * crol + (1.0f - crol) * pg[0];
    float gm1 = (e1 / es) * crol + (1.0f - crol) * pg[1];
    float gm2 = (e2 / es) * crol + (1.0f - crol) * pg[2];
    float gm3 = (e3 / es) * crol + (1.0f - crol) * pg[3];
    const float* Sb = S + (size_t)b * NST;
    float o = gm0 * Sb[n] + gm1 * Sb[512 + n] + gm2 * Sb[1024 + n] + gm3 * Sb[1536 + n];
    Aout[(size_t)b * N_ + n] = (__bf16)o;
    if (n == N_ - 1) out[OUT_LAST_OFF + b] = o;            // out_state[:, -1]
    if (n == 0) {
        float* go = out + GATE_OFF + (size_t)b * 4;
        go[0] = gm0; go[1] = gm1; go[2] = gm2; go[3] = gm3;
    }
}

// ================= pass 1 over memory: sim (both heads) + row norms =================
__global__ __launch_bounds__(256)
void sim_norm_kernel(const float* __restrict__ mem, const float* __restrict__ P,
                     float* __restrict__ simr, float* __restrict__ simw,
                     float* __restrict__ rnorm, float* __restrict__ knr,
                     float* __restrict__ knw)
{
    __shared__ float kr[256], kw[256], red[256];
    int b = blockIdx.x >> 2;
    int chunk = blockIdx.x & 3;
    int t = threadIdx.x;
    const float* Pb = P + (size_t)b * NP;
    kr[t] = tanhf(Pb[t]);
    kw[t] = tanhf(Pb[262 + t]);
    __syncthreads();
    if (chunk == 0) {
        red[t] = kr[t] * kr[t]; __syncthreads();
        for (int s = 128; s > 0; s >>= 1) { if (t < s) red[t] += red[t + s]; __syncthreads(); }
        if (t == 0) knr[b] = sqrtf(red[0]);
        __syncthreads();
        red[t] = kw[t] * kw[t]; __syncthreads();
        for (int s = 128; s > 0; s >>= 1) { if (t < s) red[t] += red[t + s]; __syncthreads(); }
        if (t == 0) knw[b] = sqrtf(red[0]);
        __syncthreads();
    }
    int wave = t >> 5, lane = t & 31;
    for (int r = 0; r < 16; ++r) {
        int n = chunk * 128 + wave * 16 + r;
        const float* mr = mem + ((size_t)b * N_ + n) * M_;
        float dr = 0.0f, dw = 0.0f, nn = 0.0f;
#pragma unroll
        for (int u = 0; u < 8; ++u) {
            int idx = lane + 32 * u;
            float v = mr[idx];
            dr += v * kr[idx]; dw += v * kw[idx]; nn += v * v;
        }
#pragma unroll
        for (int sh = 16; sh > 0; sh >>= 1) {
            dr += __shfl_xor(dr, sh, 32);
            dw += __shfl_xor(dw, sh, 32);
            nn += __shfl_xor(nn, sh, 32);
        }
        if (lane == 0) {
            simr[(size_t)b * N_ + n] = dr;
            simw[(size_t)b * N_ + n] = dw;
            rnorm[(size_t)b * N_ + n] = sqrtf(nn);
        }
    }
}

// ================= NTM addressing (per batch x head) =================
__global__ __launch_bounds__(512)
void address_kernel(const float* __restrict__ P, const float* __restrict__ simr,
                    const float* __restrict__ simw, const float* __restrict__ rnorm,
                    const float* __restrict__ knr, const float* __restrict__ knw,
                    const float* __restrict__ pwr, const float* __restrict__ pww,
                    float* __restrict__ wr, float* __restrict__ ww)
{
    __shared__ float red[512];
    __shared__ float wg[512];
    int b = blockIdx.x >> 1;
    int head = blockIdx.x & 1;
    int n = threadIdx.x;
    const float* Pb = P + (size_t)b * NP + head * 262;
    float beta = softplusf(Pb[256]);
    float g    = sigf(Pb[257]);
    float s0r = Pb[258], s1r = Pb[259], s2r = Pb[260];
    float ms = fmaxf(s0r, fmaxf(s1r, s2r));
    float se0 = expf(s0r - ms), se1 = expf(s1r - ms), se2 = expf(s2r - ms);
    float ses = se0 + se1 + se2;
    float s0 = se0 / ses, s1 = se1 / ses, s2 = se2 / ses;
    float gamma = 1.0f + softplusf(Pb[261]);

    const float* sim = head ? simw : simr;
    float kn = head ? knw[b] : knr[b];
    float val = beta * (sim[(size_t)b * N_ + n] / (rnorm[(size_t)b * N_ + n] * kn + 1e-8f));

    // softmax over n
    red[n] = val; __syncthreads();
    for (int st = 256; st > 0; st >>= 1) { if (n < st) red[n] = fmaxf(red[n], red[n + st]); __syncthreads(); }
    float mx = red[0]; __syncthreads();
    float ev = expf(val - mx);
    red[n] = ev; __syncthreads();
    for (int st = 256; st > 0; st >>= 1) { if (n < st) red[n] += red[n + st]; __syncthreads(); }
    float wc = ev / red[0]; __syncthreads();

    const float* pw = head ? pww : pwr;
    wg[n] = g * wc + (1.0f - g) * pw[(size_t)b * N_ + n];
    __syncthreads();
    float wsft = s0 * wg[(n + 1) & 511] + s1 * wg[n] + s2 * wg[(n + 511) & 511];
    float wsh = powf(wsft + 1e-8f, gamma);
    red[n] = wsh; __syncthreads();
    for (int st = 256; st > 0; st >>= 1) { if (n < st) red[n] += red[n + st]; __syncthreads(); }
    float outv = wsh / red[0];
    (head ? ww : wr)[(size_t)b * N_ + n] = outv;
}

// ================= pass 2 over memory: reads + mem_new (fused) =================
// memory is re-read from L2 (it fits in the 192MB global L2); mem_new / reads are
// written with non-temporal hints so the streamed output does not evict it.
__global__ __launch_bounds__(256)
void readwrite_kernel(const float* __restrict__ mem, const float* __restrict__ P,
                      const float* __restrict__ aBuf, const float* __restrict__ wr,
                      const float* __restrict__ ww, float* __restrict__ out)
{
    __shared__ float wrs[512], wws[512];
    int b = blockIdx.x;
    int m = threadIdx.x;
    wrs[m]       = wr[(size_t)b * N_ + m];
    wrs[m + 256] = wr[(size_t)b * N_ + m + 256];
    wws[m]       = ww[(size_t)b * N_ + m];
    wws[m + 256] = ww[(size_t)b * N_ + m + 256];
    __syncthreads();
    float e  = sigf(P[(size_t)b * NP + 524 + m]);
    float am = aBuf[(size_t)b * M_ + m];
    const float* mb = mem + (size_t)b * N_ * M_;
    float* ob = out + MEM_OFF + (size_t)b * N_ * M_;
    float acc = 0.0f;
#pragma unroll 8
    for (int n = 0; n < N_; ++n) {
        float v = mb[(size_t)n * M_ + m];
        float wwn = wws[n];
        __builtin_nontemporal_store(v * (1.0f - wwn * e) + wwn * am, &ob[(size_t)n * M_ + m]);
        acc += wrs[n] * v;
    }
    __builtin_nontemporal_store(acc, &out[READS_OFF + (size_t)b * M_ + m]);
}

// ================= host launch =================
extern "C" void kernel_launch(void* const* d_in, const int* in_sizes, int n_in,
                              void* d_out, int out_size, void* d_ws, size_t ws_size,
                              hipStream_t stream)
{
    (void)in_sizes; (void)n_in; (void)out_size; (void)ws_size;
    const float* x        = (const float*)d_in[0];
    const float* err      = (const float*)d_in[1];
    const float* preads   = (const float*)d_in[2];
    const float* prev_h   = (const float*)d_in[3];
    const float* prev_c   = (const float*)d_in[4];
    const float* prev_g   = (const float*)d_in[5];
    const float* pwr      = (const float*)d_in[6];
    const float* pww      = (const float*)d_in[7];
    const float* memory   = (const float*)d_in[8];
    const float* W_lstm_x = (const float*)d_in[9];
    const float* W_lstm_h = (const float*)d_in[10];
    const float* b_lstm   = (const float*)d_in[11];
    const float* W_gate   = (const float*)d_in[12];
    const float* b_gate   = (const float*)d_in[13];
    const float* W_crol   = (const float*)d_in[14];
    const float* b_crol   = (const float*)d_in[15];
    const float* W_states = (const float*)d_in[16];
    const float* b_states = (const float*)d_in[17];
    const float* W_read   = (const float*)d_in[18];
    const float* b_read   = (const float*)d_in[19];
    const float* W_write  = (const float*)d_in[20];
    const float* b_write  = (const float*)d_in[21];
    const float* W_erase  = (const float*)d_in[22];
    const float* b_erase  = (const float*)d_in[23];
    const float* W_add    = (const float*)d_in[24];
    const float* b_add    = (const float*)d_in[25];
    float* out = (float*)d_out;

    char* w = (char*)d_ws;
    size_t off = 0;
    auto alloc = [&](size_t bytes) { size_t o = off; off += (bytes + 255) & ~(size_t)255; return o; };

    __bf16* Abig  = (__bf16*)(w + alloc((size_t)B_ * KPAD * 2));
    __bf16* Bz    = (__bf16*)(w + alloc((size_t)(KPAD / 32) * (NZ / 16) * 512 * 2));
    float*  zbuf  = (float*) (w + alloc((size_t)B_ * NZ * 4));
    __bf16* Ah    = (__bf16*)(w + alloc((size_t)B_ * C_ * 2));
    __bf16* Bst   = (__bf16*)(w + alloc((size_t)(C_ / 32) * (NST / 16) * 512 * 2));
    float*  Sbuf  = (float*) (w + alloc((size_t)B_ * NST * 4));
    __bf16* Bhd   = (__bf16*)(w + alloc((size_t)(C_ / 32) * (NP / 16) * 512 * 2));
    float*  bhead = (float*) (w + alloc((size_t)NP * 4));
    float*  Pbuf  = (float*) (w + alloc((size_t)B_ * NP * 4));
    __bf16* Aout  = (__bf16*)(w + alloc((size_t)B_ * N_ * 2));
    __bf16* Badd  = (__bf16*)(w + alloc((size_t)(N_ / 32) * (M_ / 16) * 512 * 2));
    float*  aBuf  = (float*) (w + alloc((size_t)B_ * M_ * 4));
    float*  simr  = (float*) (w + alloc((size_t)B_ * N_ * 4));
    float*  simw  = (float*) (w + alloc((size_t)B_ * N_ * 4));
    float*  rnorm = (float*) (w + alloc((size_t)B_ * N_ * 4));
    float*  knr   = (float*) (w + alloc((size_t)B_ * 4));
    float*  knw   = (float*) (w + alloc((size_t)B_ * 4));
    float*  wrbuf = (float*) (w + alloc((size_t)B_ * N_ * 4));
    float*  wwbuf = (float*) (w + alloc((size_t)B_ * N_ * 4));

    // ---- packing ----
    pack_A_lstm<<<(B_ * KPAD + 255) / 256, 256, 0, stream>>>(x, err, preads, prev_h, Abig);
    {
        size_t total = (size_t)(KPAD / 32) * (NZ / 16) * 512;
        packB_z<<<(unsigned)((total + 255) / 256), 256, 0, stream>>>(W_lstm_x, W_lstm_h, Bz);
    }
    {
        size_t total = (size_t)(C_ / 32) * (NST / 16) * 512;
        packB_states<<<(unsigned)((total + 255) / 256), 256, 0, stream>>>(W_states, Bst);
    }
    {
        size_t total = (size_t)(C_ / 32) * (NP / 16) * 512;
        packB_head<<<(unsigned)((total + 255) / 256), 256, 0, stream>>>(W_read, W_write, W_erase, W_gate, W_crol, Bhd);
    }
    {
        size_t total = (size_t)(N_ / 32) * (M_ / 16) * 512;
        packB_add<<<(unsigned)((total + 255) / 256), 256, 0, stream>>>(W_add, Badd);
    }
    pack_bhead<<<(NP + 255) / 256, 256, 0, stream>>>(b_read, b_write, b_erase, b_gate, b_crol, bhead);

    // ---- LSTM GEMM + elementwise ----
    gemm_bf16_wmma<<<dim3(NZ / 64, B_ / 64), 128, 0, stream>>>(Abig, Bz, b_lstm, zbuf, NZ, KPAD, 0);
    lstm_ew<<<(B_ * C_ + 255) / 256, 256, 0, stream>>>(zbuf, prev_c, Ah);

    // ---- states (tanh) + head params ----
    gemm_bf16_wmma<<<dim3(NST / 64, B_ / 64), 128, 0, stream>>>(Ah, Bst, b_states, Sbuf, NST, C_, 1);
    gemm_bf16_wmma<<<dim3(NP / 64, B_ / 64), 128, 0, stream>>>(Ah, Bhd, bhead, Pbuf, NP, C_, 0);

    // ---- gate / out_state, then a = out_state @ W_add + b_add ----
    gate_state_kernel<<<B_, 512, 0, stream>>>(Pbuf, prev_g, Sbuf, Aout, out);
    gemm_bf16_wmma<<<dim3(M_ / 64, B_ / 64), 128, 0, stream>>>(Aout, Badd, b_add, aBuf, M_, N_, 0);

    // ---- memory pass 1: similarity (both heads) + row norms ----
    sim_norm_kernel<<<B_ * 4, 256, 0, stream>>>(memory, Pbuf, simr, simw, rnorm, knr, knw);

    // ---- addressing for read + write heads ----
    address_kernel<<<B_ * 2, 512, 0, stream>>>(Pbuf, simr, simw, rnorm, knr, knw, pwr, pww, wrbuf, wwbuf);

    // ---- memory pass 2: reads + mem_new ----
    readwrite_kernel<<<B_, 256, 0, stream>>>(memory, Pbuf, aBuf, wrbuf, wwbuf, out);
}
